// LSTM_37271726195285
// MI455X (gfx1250) — matrix operand order repaired and checked by reading, not compiled, not run on hardware
//
#include <hip/hip_runtime.h>

typedef __attribute__((ext_vector_type(16))) _Float16 v16h;
typedef __attribute__((ext_vector_type(8)))  _Float16 v8h;
typedef __attribute__((ext_vector_type(8)))  float    v8f;

#define HID   48
#define TSTEP 2048
#define KPAD  64      // K padded to 2 WMMA K-chunks of 32
#define G4    192     // 4*HID gate width

// gfx1250 has V_TANH_F32 (transcendental, single instruction)
__device__ __forceinline__ float fast_tanh(float x) {
  float r;
  asm("v_tanh_f32 %0, %1" : "=v"(r) : "v"(x));
  return r;
}
__device__ __forceinline__ float fast_sigmoid(float x) {
  return __builtin_fmaf(0.5f, fast_tanh(0.5f * x), 0.5f);
}

// Pack three [192,48] f32 matrices into f16 [192,64] rows (zero-padded K),
// matching the WMMA B-operand lane layout (lane = column, contiguous K halfs).
__global__ void pack_weights(const float* __restrict__ w_hh1,
                             const float* __restrict__ w_ih2,
                             const float* __restrict__ w_hh2,
                             _Float16* __restrict__ out) {
  int idx = blockIdx.x * blockDim.x + threadIdx.x;
  if (idx >= 3 * G4 * KPAD) return;
  int mat = idx / (G4 * KPAD);
  int rem = idx % (G4 * KPAD);
  int n = rem / KPAD, k = rem % KPAD;
  const float* src = (mat == 0) ? w_hh1 : (mat == 1 ? w_ih2 : w_hh2);
  out[idx] = (k < HID) ? (_Float16)src[n * HID + k] : (_Float16)0.0f;
}

__global__ __launch_bounds__(256, 1)
void lstm3_fused(const float* __restrict__ x,        // [256][2048]
                 const float* __restrict__ w_ih1,    // [192][1]
                 const float* __restrict__ b_ih1, const float* __restrict__ b_hh1,
                 const float* __restrict__ b_ih2, const float* __restrict__ b_hh2,
                 const float* __restrict__ w_lin,    // [1][48]
                 const float* __restrict__ b_lin,    // [1]
                 const _Float16* __restrict__ wpk,   // [3][192][64] f16: w_hh1,w_ih2,w_hh2
                 float* __restrict__ out) {          // [256]
  __shared__ __align__(32) _Float16 hA[3][16 * KPAD]; // raw h per layer (WMMA A layout, K-padded)
  __shared__ __align__(32) _Float16 hT[3][16 * KPAD]; // tanh(h) per layer (fed to next layer)
  __shared__ float gRec[4][16][HID];                  // recurrent-matmul gate partials
  __shared__ float gIn [4][16][HID];                  // input-proj (+bias) gate partials
  __shared__ float xs[16];

  const int tid   = threadIdx.x;
  const int lane  = tid & 31;
  const int wid   = tid >> 5;        // 8 waves
  const int gate  = wid & 3;         // i,f,g,o
  const bool recw = wid < 4;         // waves 0-3: recurrent matmul; 4-7: input matmul
  const int bbase = blockIdx.x * 16; // batch tile (16 WGs x 16 rows)

  // zero activation buffers (also gives h0 = 0 and zeroed K padding)
  for (int i = tid; i < 3 * 16 * KPAD; i += 256) {
    (&hA[0][0])[i] = (_Float16)0;
    (&hT[0][0])[i] = (_Float16)0;
  }

  // ---- register-resident weights, in WMMA B-operand layout ----
  const _Float16* whh1p = wpk;
  const _Float16* wih2p = wpk + G4 * KPAD;
  const _Float16* whh2p = wpk + 2 * G4 * KPAD;
  const int lcol = lane & 15;
  const int koff = (lane >> 4) * 16;            // B: lanes 0-15 K 0..15, lanes 16-31 K 16..31
  v16h wrec[2][3][2];                           // [whh1|whh2][N-tile][K-chunk]
  v16h win [3][2];                              // w_ih2
  float wih1v[3], biasv1[3], biasv2[3];
  if (recw) {
#pragma unroll
    for (int tl = 0; tl < 3; ++tl)
#pragma unroll
      for (int kc = 0; kc < 2; ++kc) {
        int n = gate * HID + tl * 16 + lcol;
        wrec[0][tl][kc] = *(const v16h*)(whh1p + n * KPAD + kc * 32 + koff);
        wrec[1][tl][kc] = *(const v16h*)(whh2p + n * KPAD + kc * 32 + koff);
      }
  } else {
#pragma unroll
    for (int tl = 0; tl < 3; ++tl) {
      int n = gate * HID + tl * 16 + lcol;
#pragma unroll
      for (int kc = 0; kc < 2; ++kc)
        win[tl][kc] = *(const v16h*)(wih2p + n * KPAD + kc * 32 + koff);
      wih1v[tl]  = w_ih1[n];
      biasv1[tl] = b_ih1[n] + b_hh1[n];
      biasv2[tl] = b_ih2[n] + b_hh2[n];
    }
  }

  // elementwise assignment: 768 cells/layer over 256 threads
  int em[3], ek[3];
#pragma unroll
  for (int r = 0; r < 3; ++r) { int e = tid + 256 * r; em[r] = e / HID; ek[r] = e % HID; }
  float cst[3][3] = {};   // cell state [layer][r]

  // A operand (16x32 f16, lane = row, contiguous K 16h..16h+15 per half-wave)
  auto loadA = [&](const _Float16* base, int kc) -> v16h {
    const _Float16* p = base + (lane & 15) * KPAD + kc * 32 + ((lane >> 4) * 16);
    v8h lo = *(const v8h*)p;
    v8h hi = *(const v8h*)(p + 8);
    return __builtin_shufflevector(lo, hi, 0,1,2,3,4,5,6,7,8,9,10,11,12,13,14,15);
  };
  // scatter f32 accumulator (VGPR j: row = j + 8*(lane>>4), col = lane&15)
  auto storeG = [&](float (*dst)[16][HID], const v8f* acc) {
    int rb = (lane >> 4) * 8;
#pragma unroll
    for (int tl = 0; tl < 3; ++tl)
#pragma unroll
      for (int j = 0; j < 8; ++j)
        dst[gate][rb + j][tl * 16 + lcol] = acc[tl][j];
  };
  auto elementwise = [&](int l) {
#pragma unroll
    for (int r = 0; r < 3; ++r) {
      int m = em[r], k = ek[r];
      float vi = gRec[0][m][k] + gIn[0][m][k];
      float vf = gRec[1][m][k] + gIn[1][m][k];
      float vg = gRec[2][m][k] + gIn[2][m][k];
      float vo = gRec[3][m][k] + gIn[3][m][k];
      float c = fast_sigmoid(vf) * cst[l][r] + fast_sigmoid(vi) * fast_tanh(vg);
      cst[l][r] = c;
      float h = fast_sigmoid(vo) * fast_tanh(c);
      hA[l][m * KPAD + k] = (_Float16)h;             // recurrent input next step
      hT[l][m * KPAD + k] = (_Float16)fast_tanh(h);  // layer output (extra tanh)
    }
  };

  for (int t = 0; t < TSTEP; ++t) {
    if (tid < 16) xs[tid] = x[(bbase + tid) * TSTEP + t];
    __syncthreads();

    // ---------- layer 0: gates = x*w_ih1 + bias + h1 @ w_hh1^T ----------
    {
      v8f acc[3] = {};
      if (recw) {
#pragma unroll
        for (int kc = 0; kc < 2; ++kc) {
          v16h a = loadA(&hA[0][0], kc);
#pragma unroll
          for (int tl = 0; tl < 3; ++tl)
            acc[tl] = __builtin_amdgcn_wmma_f32_16x16x32_f16(
                false, a, false, wrec[0][tl][kc], (short)0, acc[tl], false, false);
        }
        storeG(gRec, acc);
      } else {
        int rb = (lane >> 4) * 8;
#pragma unroll
        for (int tl = 0; tl < 3; ++tl)
#pragma unroll
          for (int j = 0; j < 8; ++j)
            acc[tl][j] = __builtin_fmaf(xs[rb + j], wih1v[tl], biasv1[tl]);
        storeG(gIn, acc);
      }
    }
    __syncthreads();
    elementwise(0);
    __syncthreads();

    // ---------- layers 1,2: gates = tanh(h_prev) @ w_ih2^T + bias + h_l @ w_hh2^T ----------
#pragma unroll
    for (int l = 1; l < 3; ++l) {
      v8f acc[3] = {};
      if (recw) {
#pragma unroll
        for (int kc = 0; kc < 2; ++kc) {
          v16h a = loadA(&hA[l][0], kc);
#pragma unroll
          for (int tl = 0; tl < 3; ++tl)
            acc[tl] = __builtin_amdgcn_wmma_f32_16x16x32_f16(
                false, a, false, wrec[1][tl][kc], (short)0, acc[tl], false, false);
        }
        storeG(gRec, acc);
      } else {
#pragma unroll
        for (int tl = 0; tl < 3; ++tl)
#pragma unroll
          for (int j = 0; j < 8; ++j) acc[tl][j] = biasv2[tl];
#pragma unroll
        for (int kc = 0; kc < 2; ++kc) {
          v16h a = loadA(&hT[l - 1][0], kc);
#pragma unroll
          for (int tl = 0; tl < 3; ++tl)
            acc[tl] = __builtin_amdgcn_wmma_f32_16x16x32_f16(
                false, a, false, win[tl][kc], (short)0, acc[tl], false, false);
        }
        storeG(gIn, acc);
      }
      __syncthreads();
      elementwise(l);
      __syncthreads();
    }
  }

  // final linear on last timestep: out = tanh(h3_last) @ w_lin^T + b_lin
  if (tid < 16) {
    float s = b_lin[0];
#pragma unroll
    for (int k = 0; k < HID; ++k)
      s += (float)hT[2][tid * KPAD + k] * w_lin[k];
    out[bbase + tid] = s;
  }
}

extern "C" void kernel_launch(void* const* d_in, const int* in_sizes, int n_in,
                              void* d_out, int out_size, void* d_ws, size_t ws_size,
                              hipStream_t stream) {
  const float* x     = (const float*)d_in[0];
  const float* w_ih1 = (const float*)d_in[1];
  const float* w_hh1 = (const float*)d_in[2];
  const float* b_ih1 = (const float*)d_in[3];
  const float* b_hh1 = (const float*)d_in[4];
  const float* w_ih2 = (const float*)d_in[5];
  const float* w_hh2 = (const float*)d_in[6];
  const float* b_ih2 = (const float*)d_in[7];
  const float* b_hh2 = (const float*)d_in[8];
  const float* w_lin = (const float*)d_in[9];
  const float* b_lin = (const float*)d_in[10];

  _Float16* wpk = (_Float16*)d_ws;                 // 3*192*64*2 = 73,728 bytes
  int total = 3 * G4 * KPAD;
  pack_weights<<<(total + 255) / 256, 256, 0, stream>>>(w_hh1, w_ih2, w_hh2, wpk);
  lstm3_fused<<<16, 256, 0, stream>>>(x, w_ih1, b_ih1, b_hh1, b_ih2, b_hh2,
                                      w_lin, b_lin, wpk, (float*)d_out);
}
